// RecurrentNet_67233418052080
// MI455X (gfx1250) — matrix operand order repaired and checked by reading, not compiled, not run on hardware
//
#include <hip/hip_runtime.h>

// ---------------------------------------------------------------------------
// Fused 2-layer GRU (B=256, T=2048, I=64, H=128) + FC(128->10) on last step.
// 16 workgroups (16 batch rows each) x 8 waves (16 hidden cols each).
// Weights live in VGPRs as WMMA B-operands (bf16); bf16 h-state in LDS for
// WMMA A-operands (double-buffered); f32 h_prev carried in registers (each
// lane's C/D elements at step t are the same (M,N) it produced at t-1).
// Gate nonlinearities use the CDNA5 TRANS unit (v_tanh_f32 / v_rcp_f32),
// avoiding the IEEE division expansion seen in round 1.
// ---------------------------------------------------------------------------

typedef __attribute__((ext_vector_type(16))) __bf16 v16bf;
typedef __attribute__((ext_vector_type(8)))  float  v8f;

#define B_TOT 256
#define T_LEN 2048
#define I_DIM 64
#define H_DIM 128
#define O_DIM 10
#define BT    16      // batch tile per workgroup (WMMA M)
#define NW    8       // waves per block (8 * 16 cols = H)

__device__ __forceinline__ v8f splat8(float v) {
  v8f r;
#pragma unroll
  for (int i = 0; i < 8; ++i) r[i] = v;
  return r;
}

__device__ __forceinline__ float fast_tanh(float v) {
#if __has_builtin(__builtin_amdgcn_tanhf)
  return __builtin_amdgcn_tanhf(v);            // v_tanh_f32 (CDNA5 TRANS op)
#else
  // tanh(v) = 2/(1+exp(-2v)) - 1, with raw v_rcp_f32 (no div expansion)
  const float t = __builtin_amdgcn_rcpf(1.0f + __expf(-2.0f * v));
  return fmaf(2.0f, t, -1.0f);
#endif
}

__device__ __forceinline__ float fast_sigmoid(float v) {
#if __has_builtin(__builtin_amdgcn_tanhf)
  // sigmoid(v) = 0.5*tanh(v/2) + 0.5 : 1 TRANS + 2 VALU
  return fmaf(0.5f, __builtin_amdgcn_tanhf(0.5f * v), 0.5f);
#else
  return __builtin_amdgcn_rcpf(1.0f + __expf(-v));
#endif
}

// WMMA B-operand (32x16 bf16, K x N) from row-major fp32 W[3H][Kin].
// B layout: lane L -> N = L%16; elems 0..7 -> K = kbase+e, elems 8..15 -> K = kbase+16+(e-8).
__device__ __forceinline__ v16bf load_b_w(const float* __restrict__ W, int Kin,
                                          int row, int kbase) {
  union { float4 f4[4]; float f[16]; } u;
  const float4* p = (const float4*)(W + (size_t)row * Kin + kbase);
  u.f4[0] = p[0]; u.f4[1] = p[1]; u.f4[2] = p[4]; u.f4[3] = p[5];
  v16bf v;
#pragma unroll
  for (int i = 0; i < 16; ++i) v[i] = (__bf16)u.f[i];
  return v;
}

// WMMA A-operand (16x32 bf16, M x K) from LDS row-major bf16 tile.
// A layout: lane L -> M = L%16; elems 0..7 -> K = kbase+e,
// elems 8..15 -> K = kbase+16+(e-8), where kbase = kt*32 + (L>=16?8:0).
__device__ __forceinline__ v16bf load_a(const __bf16* p, int kbase) {
  union { v16bf v; uint4 u[2]; } r;
  r.u[0] = *(const uint4*)(p + kbase);
  r.u[1] = *(const uint4*)(p + kbase + 16);
  return r.v;
}

__device__ __forceinline__ v8f wmma_bf16(v16bf a, v16bf b, v8f c) {
  return __builtin_amdgcn_wmma_f32_16x16x32_bf16(false, a, false, b,
                                                 (short)0, c, false, false);
}

__global__ void __launch_bounds__(256, 1)
gru2_fused_kernel(const float* __restrict__ x,
                  const float* __restrict__ Wih0, const float* __restrict__ Whh0,
                  const float* __restrict__ bih0, const float* __restrict__ bhh0,
                  const float* __restrict__ Wih1, const float* __restrict__ Whh1,
                  const float* __restrict__ bih1, const float* __restrict__ bhh1,
                  const float* __restrict__ fcW,  const float* __restrict__ fcb,
                  float* __restrict__ out) {
  __shared__ __align__(16) __bf16 xbf[2][BT][I_DIM];   // staged x_t (bf16)
  __shared__ __align__(16) __bf16 h0b[2][BT][H_DIM];   // h0 bf16 (WMMA A), dbl-buf
  __shared__ __align__(16) __bf16 h1b[2][BT][H_DIM];   // h1 bf16 (WMMA A), dbl-buf
  __shared__ __align__(16) float  h1f[BT][H_DIM];      // h1 f32, FC only (post-loop)

  const int tid   = threadIdx.x;
  const int lane  = tid & 31;
  const int wave  = tid >> 5;          // 0..7
  const int ln16  = lane & 15;
  const int hi    = lane >> 4;         // 0 or 1 (half-wave)
  const int cb    = wave * 16;         // this wave's hidden-column base
  const int j     = cb + ln16;         // this lane's hidden column (N)
  const int bbase = blockIdx.x * BT;

  // ---------------- weight preload into VGPRs (B-operands, bf16) -----------
  const int koffB = hi * 8;            // K offset within 32-wide tile (B layout)
  v16bf w_ih0[3][2], w_hh0[3][4], w_ih1[3][4], w_hh1[3][4];
#pragma unroll
  for (int g = 0; g < 3; ++g) {        // gate: 0=r, 1=z, 2=n
    const int row = g * H_DIM + j;     // W row (output gate unit) for this lane
#pragma unroll
    for (int kt = 0; kt < 2; ++kt)
      w_ih0[g][kt] = load_b_w(Wih0, I_DIM, row, kt * 32 + koffB);
#pragma unroll
    for (int kt = 0; kt < 4; ++kt) {
      w_hh0[g][kt] = load_b_w(Whh0, H_DIM, row, kt * 32 + koffB);
      w_ih1[g][kt] = load_b_w(Wih1, H_DIM, row, kt * 32 + koffB);
      w_hh1[g][kt] = load_b_w(Whh1, H_DIM, row, kt * 32 + koffB);
    }
  }

  // Per-lane gate biases (column j fixed per lane -> splat as WMMA C init).
  const float b0r  = bih0[j] + bhh0[j];
  const float b0z  = bih0[H_DIM + j] + bhh0[H_DIM + j];
  const float b0nx = bih0[2 * H_DIM + j];
  const float b0nh = bhh0[2 * H_DIM + j];
  const float b1r  = bih1[j] + bhh1[j];
  const float b1z  = bih1[H_DIM + j] + bhh1[H_DIM + j];
  const float b1nx = bih1[2 * H_DIM + j];
  const float b1nh = bhh1[2 * H_DIM + j];

  // ---------------- h state init ------------------------------------------
  for (int idx = tid; idx < BT * H_DIM; idx += 256) {
    const int m = idx >> 7, k = idx & 127;
    h0b[1][m][k] = (__bf16)0.0f;
    h1b[1][m][k] = (__bf16)0.0f;
  }
  float hp0[8], hp1[8];                // per-lane f32 h_prev (M = e+8*hi, N = j)
#pragma unroll
  for (int e = 0; e < 8; ++e) { hp0[e] = 0.0f; hp1[e] = 0.0f; }

  // ---------------- x prefetch (register double-buffer) --------------------
  const int xrow = tid >> 4;                 // 0..15 (batch row in tile)
  const int xc4  = (tid & 15) * 4;           // float4 column
  const float* xbase =
      x + (size_t)(bbase + xrow) * T_LEN * I_DIM + xc4;
  float4 xr = *(const float4*)xbase;         // t = 0

  const int koffA = hi * 8;                  // K offset for A-operand chunks
  const int M     = ln16;                    // A row = batch row (lane % 16)

  for (int t = 0; t < T_LEN; ++t) {
    const int pb = t & 1;                    // buffer produced this step
    const int rb = pb ^ 1;                   // buffer holding t-1 state

    // Stage x_t to LDS (bf16) and kick off the t+1 global prefetch.
    {
      __bf16* xd = &xbf[pb][xrow][xc4];
      xd[0] = (__bf16)xr.x; xd[1] = (__bf16)xr.y;
      xd[2] = (__bf16)xr.z; xd[3] = (__bf16)xr.w;
      if (t + 1 < T_LEN)
        xr = *(const float4*)(xbase + (size_t)(t + 1) * I_DIM);
    }
    __syncthreads();   // x_t staged; h0b[pb]/h1b[pb] free to overwrite

    // ======================= layer 0 =======================
    v8f cr  = splat8(b0r);
    v8f cz  = splat8(b0z);
    v8f cnx = splat8(b0nx);   // x-part of n gate (tanh(xn + r*hn) needs it split)
    v8f cnh = splat8(b0nh);   // h-part of n gate
#pragma unroll
    for (int kt = 0; kt < 2; ++kt) {         // K = 64 over x_t
      v16bf a = load_a(&xbf[pb][M][0], kt * 32 + koffA);
      cr  = wmma_bf16(a, w_ih0[0][kt], cr);
      cz  = wmma_bf16(a, w_ih0[1][kt], cz);
      cnx = wmma_bf16(a, w_ih0[2][kt], cnx);
    }
#pragma unroll
    for (int kt = 0; kt < 4; ++kt) {         // K = 128 over h0(t-1)
      v16bf a = load_a(&h0b[rb][M][0], kt * 32 + koffA);
      cr  = wmma_bf16(a, w_hh0[0][kt], cr);
      cz  = wmma_bf16(a, w_hh0[1][kt], cz);
      cnh = wmma_bf16(a, w_hh0[2][kt], cnh);
    }
#pragma unroll
    for (int e = 0; e < 8; ++e) {            // C/D layout: M = e + 8*hi, N = j
      const int m = e + hi * 8;
      const float r  = fast_sigmoid(cr[e]);
      const float z  = fast_sigmoid(cz[e]);
      const float n  = fast_tanh(cnx[e] + r * cnh[e]);
      const float hn = n + z * (hp0[e] - n); // (1-z)*n + z*h_prev
      hp0[e] = hn;
      h0b[pb][m][j] = (__bf16)hn;
    }
    __syncthreads();   // h0(t) visible to all waves

    // ======================= layer 1 =======================
    cr  = splat8(b1r);
    cz  = splat8(b1z);
    cnx = splat8(b1nx);
    cnh = splat8(b1nh);
#pragma unroll
    for (int kt = 0; kt < 4; ++kt) {         // K = 128 over h0(t)
      v16bf a = load_a(&h0b[pb][M][0], kt * 32 + koffA);
      cr  = wmma_bf16(a, w_ih1[0][kt], cr);
      cz  = wmma_bf16(a, w_ih1[1][kt], cz);
      cnx = wmma_bf16(a, w_ih1[2][kt], cnx);
    }
#pragma unroll
    for (int kt = 0; kt < 4; ++kt) {         // K = 128 over h1(t-1)
      v16bf a = load_a(&h1b[rb][M][0], kt * 32 + koffA);
      cr  = wmma_bf16(a, w_hh1[0][kt], cr);
      cz  = wmma_bf16(a, w_hh1[1][kt], cz);
      cnh = wmma_bf16(a, w_hh1[2][kt], cnh);
    }
#pragma unroll
    for (int e = 0; e < 8; ++e) {
      const int m = e + hi * 8;
      const float r  = fast_sigmoid(cr[e]);
      const float z  = fast_sigmoid(cz[e]);
      const float n  = fast_tanh(cnx[e] + r * cnh[e]);
      const float hn = n + z * (hp1[e] - n);
      hp1[e] = hn;
      h1b[pb][m][j] = (__bf16)hn;
    }
    // next iteration's leading __syncthreads orders h1b[pb] before reuse
  }

  // ---------------- FC on h1(T-1): out = h1 @ fcW^T + fcb ------------------
  // Publish the register-resident final h1 to LDS once, then tiny FC.
#pragma unroll
  for (int e = 0; e < 8; ++e)
    h1f[e + hi * 8][j] = hp1[e];
  __syncthreads();
  if (tid < BT * O_DIM) {                    // 160 outputs per workgroup
    const int bb = tid / O_DIM;
    const int o  = tid % O_DIM;
    float acc = fcb[o];
#pragma unroll 4
    for (int k = 0; k < H_DIM; ++k)
      acc = fmaf(h1f[bb][k], fcW[o * H_DIM + k], acc);
    out[(size_t)(bbase + bb) * O_DIM + o] = acc;
  }
}

extern "C" void kernel_launch(void* const* d_in, const int* in_sizes, int n_in,
                              void* d_out, int out_size, void* d_ws, size_t ws_size,
                              hipStream_t stream) {
  (void)in_sizes; (void)n_in; (void)out_size; (void)d_ws; (void)ws_size;
  const float* x    = (const float*)d_in[0];
  const float* Wih0 = (const float*)d_in[1];
  const float* Whh0 = (const float*)d_in[2];
  const float* bih0 = (const float*)d_in[3];
  const float* bhh0 = (const float*)d_in[4];
  const float* Wih1 = (const float*)d_in[5];
  const float* Whh1 = (const float*)d_in[6];
  const float* bih1 = (const float*)d_in[7];
  const float* bhh1 = (const float*)d_in[8];
  const float* fcW  = (const float*)d_in[9];
  const float* fcb  = (const float*)d_in[10];
  float* out = (float*)d_out;

  dim3 grid(B_TOT / BT);   // 16 workgroups
  dim3 block(NW * 32);     // 256 threads = 8 waves
  gru2_fused_kernel<<<grid, block, 0, stream>>>(
      x, Wih0, Whh0, bih0, bhh0, Wih1, Whh1, bih1, bhh1, fcW, fcb, out);
}
